// TransformerBlock_3289944949306
// MI455X (gfx1250) — matrix operand order repaired and checked by reading, not compiled, not run on hardware
//
#include <hip/hip_runtime.h>
#include <math.h>
#include <stdint.h>

typedef _Float16 half_t;
typedef __attribute__((ext_vector_type(16))) _Float16 v16h;
typedef __attribute__((ext_vector_type(8)))  _Float16 v8h;
typedef __attribute__((ext_vector_type(8)))  float    v8f;

#define WMMA_F16(a, b, c) \
  __builtin_amdgcn_wmma_f32_16x16x32_f16(false, (a), false, (b), (short)0, (c), false, false)

// ---- CDNA5 async global->LDS copy (ASYNCcnt tracked, no VGPR round trip) ---
__device__ inline uint32_t lds_off(const void* p) {
  return (uint32_t)(uintptr_t)p;  // low 32 bits of flat addr = LDS byte offset
}
__device__ inline void async_b128(uint32_t dst_lds, const void* src) {
  asm volatile("global_load_async_to_lds_b128 %0, %1, off"
               :: "v"(dst_lds), "v"((uint64_t)(uintptr_t)src) : "memory");
}

// ---- fragment loaders (16B-aligned) ----------------------------------------
// A 16x32 f16: lane<16 holds K{0..7,16..23}, lane>=16 holds K{8..15,24..31}
__device__ inline v16h load_frag_a(const half_t* p) {
  v8h lo = *(const v8h*)(p);
  v8h hi = *(const v8h*)(p + 16);
  return __builtin_shufflevector(lo, hi, 0,1,2,3,4,5,6,7,8,9,10,11,12,13,14,15);
}
// B 32x16 f16: lane<16 holds K{0..7,16..23}, lane>=16 holds K{8..15,24..31}
__device__ inline v16h load_frag_b16(const half_t* p) {
  v8h lo = *(const v8h*)(p);
  v8h hi = *(const v8h*)(p + 16);
  return __builtin_shufflevector(lo, hi, 0,1,2,3,4,5,6,7,8,9,10,11,12,13,14,15);
}

// ---- one-time weight transpose + f32->f16: WT[n][k] = W[k][n] --------------
__global__ __launch_bounds__(256)
void transpose_cvt(const float* __restrict__ W, half_t* __restrict__ WT,
                   int K, int N) {
  __shared__ float t[32][33];
  const int n0 = blockIdx.x * 32, k0 = blockIdx.y * 32;
  const int lx = threadIdx.x & 31, ly = threadIdx.x >> 5;  // 32 x 8
  for (int i = 0; i < 32; i += 8)
    t[ly + i][lx] = W[(long)(k0 + ly + i) * N + n0 + lx];
  __syncthreads();
  for (int i = 0; i < 32; i += 8)
    WT[(long)(n0 + ly + i) * K + k0 + lx] = (half_t)t[lx][ly + i];
}

// ---- LayerNorm: one row (D=512) per 256-thread block, out f16 --------------
__global__ __launch_bounds__(256)
void ln_kernel(const float* __restrict__ in, const float* __restrict__ g,
               const float* __restrict__ b, half_t* __restrict__ out) {
  const int row = blockIdx.x;
  const int tid = threadIdx.x;
  float2 v = ((const float2*)(in + (long)row * 512))[tid];
  float s = v.x + v.y;
  float ss = v.x * v.x + v.y * v.y;
  for (int m = 1; m < 32; m <<= 1) {
    s  += __shfl_xor(s,  m, 32);
    ss += __shfl_xor(ss, m, 32);
  }
  __shared__ float as_[8], ass_[8];
  int w = tid >> 5, l = tid & 31;
  if (l == 0) { as_[w] = s; ass_[w] = ss; }
  __syncthreads();
  s = 0.f; ss = 0.f;
  for (int i = 0; i < 8; i++) { s += as_[i]; ss += ass_[i]; }
  float mu   = s * (1.0f / 512.0f);
  float var  = ss * (1.0f / 512.0f) - mu * mu;
  float rstd = rsqrtf(var + 1e-5f);
  int c = tid * 2;
  out[(long)row * 512 + c]     = (half_t)((v.x - mu) * rstd * g[c]     + b[c]);
  out[(long)row * 512 + c + 1] = (half_t)((v.y - mu) * rstd * g[c + 1] + b[c + 1]);
}

// ---- WMMA GEMM: C[MxN] = A[MxK](f16) * WT[NxK](f16), async double-buffered -
// Block tile 128x64, BK=64, 256 threads = 8 waves, each wave 16x64.
// MODE 0: scatter q(*0.125) [B,H,T,HD], k [B,H,T,HD], v [B,H,HD,T]  (f16)
// MODE 1: out_f32 = resid + acc
// MODE 2: out_h   = gelu(acc + bias)
// MODE 3: out_f32 = resid + bias + acc
#define BM 128
#define BN 64
#define BK 64
#define LDW 72  // padded row stride (halfs) for both LDS tiles

template <int MODE>
__global__ __launch_bounds__(256)
void gemm_wmma(const half_t* __restrict__ A, const half_t* __restrict__ WT,
               int K, int N,
               const float* __restrict__ resid, const float* __restrict__ bias,
               float* __restrict__ outf, half_t* __restrict__ outh,
               half_t* __restrict__ oq, half_t* __restrict__ ok,
               half_t* __restrict__ ov) {
  __shared__ half_t sa[2][BM * LDW];
  __shared__ half_t sb[2][BN * LDW];
  const int tid  = threadIdx.x;
  const int wave = tid >> 5;
  const int lane = tid & 31;
  const int hlf  = lane >> 4;
  const int idx  = lane & 15;
  const int row0 = blockIdx.y * BM;
  const int col0 = blockIdx.x * BN;

  v8f acc[4] = {};

  // issue async copies for one k-stage: A 128x64 (4 chunks/thr), B 64x64 (2)
  auto issue = [&](int k0, int buf) {
    for (int i = 0; i < 4; i++) {
      int c = tid * 4 + i;            // 0..1023
      int m = c >> 3, part = c & 7;   // 8 x 16B chunks per row
      async_b128(lds_off(&sa[buf][m * LDW + part * 8]),
                 &A[(long)(row0 + m) * K + k0 + part * 8]);
    }
    for (int i = 0; i < 2; i++) {
      int c = tid * 2 + i;            // 0..511
      int n = c >> 3, part = c & 7;
      async_b128(lds_off(&sb[buf][n * LDW + part * 8]),
                 &WT[(long)(col0 + n) * K + k0 + part * 8]);
    }
  };

  issue(0, 0);
  const int nk = K / BK;
  for (int kb = 0; kb < nk; kb++) {
    const int cur = kb & 1;
    if (kb + 1 < nk) {
      issue((kb + 1) * BK, cur ^ 1);
      asm volatile("s_wait_asynccnt 6" ::: "memory");  // stage kb resident
    } else {
      asm volatile("s_wait_asynccnt 0" ::: "memory");
    }
    __syncthreads();

    const int m_base = wave * 16;
    for (int ks = 0; ks < 2; ks++) {
      v16h af = load_frag_a(&sa[cur][(m_base + idx) * LDW + ks * 32 + hlf * 8]);
      for (int j = 0; j < 4; j++) {
        v16h bf = load_frag_b16(&sb[cur][(j * 16 + idx) * LDW + ks * 32 + hlf * 8]);
        acc[j] = WMMA_F16(af, bf, acc[j]);
      }
    }
    __syncthreads();  // all waves done reading before buffer reuse
  }

  // ---- epilogue; C layout: VGPR r -> row (hlf*8 + r), col (j*16 + idx) ----
  const int mrow = row0 + wave * 16 + hlf * 8;
  for (int j = 0; j < 4; j++) {
    const int col = col0 + j * 16 + idx;
    for (int r = 0; r < 8; r++) {
      float vv = acc[j][r];
      int row = mrow + r;
      if constexpr (MODE == 0) {
        int which = col >> 9;          // 0=q 1=k 2=v
        int rem = col & 511;
        int hh = rem >> 6, hd = rem & 63;
        int bb = row >> 12, t = row & 4095;
        if (which == 0)
          oq[(((long)(bb * 8 + hh)) * 4096 + t) * 64 + hd] = (half_t)(vv * 0.125f);
        else if (which == 1)
          ok[(((long)(bb * 8 + hh)) * 4096 + t) * 64 + hd] = (half_t)vv;
        else  // v stored transposed [B,H,HD,T] so attention can async-copy it
          ov[(((long)(bb * 8 + hh)) * 64 + hd) * 4096 + t] = (half_t)vv;
      } else if constexpr (MODE == 1) {
        long di = (long)row * N + col;
        outf[di] = resid[di] + vv;
      } else if constexpr (MODE == 2) {
        float t = vv + bias[col];
        outh[(long)row * N + col] =
            (half_t)(0.5f * t * (1.0f + erff(t * 0.70710678f)));
      } else {
        long di = (long)row * N + col;
        outf[di] = resid[di] + bias[col] + vv;
      }
    }
  }
}

// ---- causal flash attention: 4 waves, 64 q-rows per block ------------------
// grid = (T/64, B*H); q,k f16 [B*H,T,64]; v f16 [B*H,64,T]; out f16 [B,T,512]
__global__ __launch_bounds__(128)
void attn_kernel(const half_t* __restrict__ q, const half_t* __restrict__ kx,
                 const half_t* __restrict__ vx, half_t* __restrict__ aout) {
  constexpr int T = 4096, HD = 64;
  __shared__ half_t sk[64 * LDW];      // [krow][hd]
  __shared__ half_t sv[64 * LDW];      // [hd][krow]  (v already transposed)
  __shared__ half_t sp[4][16 * LDW];   // wave-private P staging

  const int tid  = threadIdx.x;
  const int wave = tid >> 5;
  const int lane = tid & 31;
  const int hlf  = lane >> 4;
  const int idx  = lane & 15;
  const int qt = blockIdx.x;
  const int bh = blockIdx.y;

  const half_t* qbase = q  + ((long)bh * T + qt * 64) * HD;
  const half_t* kbase = kx + (long)bh * T * HD;
  const half_t* vbase = vx + (long)bh * HD * T;

  v16h qa[2];
  for (int ks = 0; ks < 2; ks++)
    qa[ks] = load_frag_a(qbase + (wave * 16 + idx) * HD + ks * 32 + hlf * 8);

  v8f o[4] = {};
  float mrow[8], lrow[8];
  for (int r = 0; r < 8; r++) { mrow[r] = -1e30f; lrow[r] = 0.f; }

  for (int kt = 0; kt <= qt; kt++) {   // causal: future tiles skipped
    __syncthreads();                   // previous readers done
    for (int i = 0; i < 4; i++) {      // K tile: 512 x 16B chunks / 128 thr
      int c = tid * 4 + i;
      int r2 = c >> 3, part = c & 7;
      async_b128(lds_off(&sk[r2 * LDW + part * 8]),
                 kbase + ((long)kt * 64 + r2) * HD + part * 8);
    }
    for (int i = 0; i < 4; i++) {      // V tile (transposed source)
      int c = tid * 4 + i;
      int hd = c >> 3, part = c & 7;
      async_b128(lds_off(&sv[hd * LDW + part * 8]),
                 vbase + (long)hd * T + kt * 64 + part * 8);
    }
    asm volatile("s_wait_asynccnt 0" ::: "memory");
    __syncthreads();

    // S = Q K^T  (16 q-rows x 64 k-cols per wave)
    v8f s[4] = {};
    for (int ks = 0; ks < 2; ks++)
      for (int j = 0; j < 4; j++) {
        v16h bf = load_frag_b16(&sk[(j * 16 + idx) * LDW + ks * 32 + hlf * 8]);
        s[j] = WMMA_F16(qa[ks], bf, s[j]);
      }

    if (kt == qt) {
      for (int j = 0; j < 4; j++) {
        int colh = j * 16 + idx;
        for (int r = 0; r < 8; r++)
          if (colh > wave * 16 + hlf * 8 + r) s[j][r] = -1e30f;
      }
    }

    // online softmax; row r lives in a 16-lane half -> xor-reduce over 16
    float alpha[8];
    for (int r = 0; r < 8; r++) {
      float mx = fmaxf(fmaxf(s[0][r], s[1][r]), fmaxf(s[2][r], s[3][r]));
      for (int mk = 1; mk < 16; mk <<= 1) mx = fmaxf(mx, __shfl_xor(mx, mk, 32));
      float mnew = fmaxf(mrow[r], mx);
      alpha[r] = __expf(mrow[r] - mnew);
      mrow[r] = mnew;
    }
    for (int r = 0; r < 8; r++) {
      float su = 0.f;
      for (int j = 0; j < 4; j++) {
        float p = __expf(s[j][r] - mrow[r]);
        s[j][r] = p;
        su += p;
      }
      for (int mk = 1; mk < 16; mk <<= 1) su += __shfl_xor(su, mk, 32);
      lrow[r] = lrow[r] * alpha[r] + su;
    }
    for (int jj = 0; jj < 4; jj++)
      for (int r = 0; r < 8; r++) o[jj][r] *= alpha[r];

    // C-layout -> A-layout via wave-private LDS, then O += P V
    for (int j = 0; j < 4; j++) {
      int colh = j * 16 + idx;
      for (int r = 0; r < 8; r++)
        sp[wave][(hlf * 8 + r) * LDW + colh] = (half_t)s[j][r];
    }
    asm volatile("s_wait_dscnt 0" ::: "memory");
    for (int ks = 0; ks < 2; ks++) {
      v16h pa = load_frag_a(&sp[wave][idx * LDW + ks * 32 + hlf * 8]);
      for (int jj = 0; jj < 4; jj++) {
        v16h bf = load_frag_b16(&sv[(jj * 16 + idx) * LDW + ks * 32 + hlf * 8]);
        o[jj] = WMMA_F16(pa, bf, o[jj]);
      }
    }
  }

  const int bb = bh >> 3, hh = bh & 7;
  for (int r = 0; r < 8; r++) {
    float inv = 1.0f / lrow[r];
    int t = qt * 64 + wave * 16 + hlf * 8 + r;
    half_t* dst = aout + ((long)bb * T + t) * 512 + hh * 64;
    for (int jj = 0; jj < 4; jj++)
      dst[jj * 16 + idx] = (half_t)(o[jj][r] * inv);
  }
}

// ---------------------------------------------------------------------------
extern "C" void kernel_launch(void* const* d_in, const int* in_sizes, int n_in,
                              void* d_out, int out_size, void* d_ws, size_t ws_size,
                              hipStream_t stream) {
  (void)in_sizes; (void)n_in; (void)out_size; (void)ws_size;
  const float* h    = (const float*)d_in[0];
  const float* ln1g = (const float*)d_in[1];
  const float* ln1b = (const float*)d_in[2];
  const float* wqkv = (const float*)d_in[3];
  const float* wout = (const float*)d_in[4];
  const float* ln2g = (const float*)d_in[5];
  const float* ln2b = (const float*)d_in[6];
  const float* wff1 = (const float*)d_in[7];
  const float* bff1 = (const float*)d_in[8];
  const float* wff2 = (const float*)d_in[9];
  const float* bff2 = (const float*)d_in[10];
  float* out = (float*)d_out;

  const long MT = 8192;  // B*T
  char* w = (char*)d_ws;
  half_t* x1 = (half_t*)w; w += MT * 512 * 2;
  half_t* qh = (half_t*)w; w += MT * 512 * 2;
  half_t* kh = (half_t*)w; w += MT * 512 * 2;
  half_t* vh = (half_t*)w; w += MT * 512 * 2;
  half_t* ah = (half_t*)w; w += MT * 512 * 2;
  float*  h2 = (float*)w;  w += MT * 512 * 4;
  half_t* x2 = (half_t*)w; w += MT * 512 * 2;
  half_t* ff = (half_t*)w; w += MT * 2048 * 2;
  half_t* wqkvT = (half_t*)w; w += 1536L * 512 * 2;  // [N][K]
  half_t* woutT = (half_t*)w; w += 512L  * 512 * 2;
  half_t* wff1T = (half_t*)w; w += 2048L * 512 * 2;
  half_t* wff2T = (half_t*)w; w += 512L  * 2048 * 2;

  // one-time f32->f16 weight transposes (K-contiguous for async staging)
  transpose_cvt<<<dim3(1536 / 32, 512 / 32),  256, 0, stream>>>(wqkv, wqkvT, 512, 1536);
  transpose_cvt<<<dim3(512 / 32,  512 / 32),  256, 0, stream>>>(wout, woutT, 512, 512);
  transpose_cvt<<<dim3(2048 / 32, 512 / 32),  256, 0, stream>>>(wff1, wff1T, 512, 2048);
  transpose_cvt<<<dim3(512 / 32,  2048 / 32), 256, 0, stream>>>(wff2, wff2T, 2048, 512);

  ln_kernel<<<8192, 256, 0, stream>>>(h, ln1g, ln1b, x1);

  gemm_wmma<0><<<dim3(1536 / BN, MT / BM), 256, 0, stream>>>(
      x1, wqkvT, 512, 1536, nullptr, nullptr, nullptr, nullptr, qh, kh, vh);

  attn_kernel<<<dim3(64, 16), 128, 0, stream>>>(qh, kh, vh, ah);

  gemm_wmma<1><<<dim3(512 / BN, MT / BM), 256, 0, stream>>>(
      ah, woutT, 512, 512, h, nullptr, h2, nullptr, nullptr, nullptr, nullptr);

  ln_kernel<<<8192, 256, 0, stream>>>(h2, ln2g, ln2b, x2);

  gemm_wmma<2><<<dim3(2048 / BN, MT / BM), 256, 0, stream>>>(
      x2, wff1T, 512, 2048, nullptr, bff1, nullptr, ff, nullptr, nullptr, nullptr);

  gemm_wmma<3><<<dim3(512 / BN, MT / BM), 256, 0, stream>>>(
      ff, wff2T, 2048, 512, h2, bff2, out, nullptr, nullptr, nullptr, nullptr);
}
